// MyModel_84327387889706
// MI455X (gfx1250) — compile-verified, matrix operand order checked by reading, not run on hardware
//
#include <hip/hip_runtime.h>

// Problem constants (match reference)
#define B_    64
#define S_    512
#define EMB_  1024
#define OUT_  10

typedef __attribute__((ext_vector_type(16))) __bf16 v16bf;
typedef __attribute__((ext_vector_type(8)))  float  v8f;

union Tile16 {
    uint4 u[2];
    v16bf v;
};

__device__ __forceinline__ unsigned short f2bf(float f) {
    unsigned int u = __float_as_uint(f);
    u += 0x7FFFu + ((u >> 16) & 1u);   // round-to-nearest-even
    return (unsigned short)(u >> 16);
}

// ---------------------------------------------------------------------------
// Kernel 1: convert W (fp32, row-major [n][k]) -> bf16 in workspace
// ---------------------------------------------------------------------------
__global__ void cvt_w_kernel(const float* __restrict__ W,
                             unsigned short* __restrict__ Wb) {
    int i = blockIdx.x * 256 + threadIdx.x;
    Wb[i] = f2bf(W[i]);
}

// ---------------------------------------------------------------------------
// Kernel 2: the affine recurrence. 4 blocks x 512 threads (16 waves, wave32).
// Block handles 16 batch rows; wave w owns output columns [64w, 64w+64).
// Per step: stage A=(h_prev + emb[x_i]) in LDS (bf16), then 16x1024x1024 GEMM
// via v_wmma_f32_16x16x32_bf16, W streamed from L2 as bf16.
// ---------------------------------------------------------------------------
__global__ __launch_bounds__(512, 1) void recur_kernel(
    const int*            __restrict__ x,       // (B, S)
    const float*          __restrict__ emb,     // (VOCAB, EMB)
    const unsigned short* __restrict__ Wb,      // (EMB, EMB) bf16, row n = out col
    const float*          __restrict__ bvec,    // (EMB,)
    const float*          __restrict__ Wc,      // (OUT, EMB)
    const float*          __restrict__ bc,      // (OUT,)
    float*                __restrict__ predmean)// (B, OUT)
{
    __shared__ unsigned long long smem[8192];            // 64 KB
    unsigned short* As = (unsigned short*)smem;          // 16 x 1024 bf16 (A stage)
    float*          Hs = (float*)smem;                   // 16 x 1024 f32 (after loop)

    const int tid    = threadIdx.x;
    const int w      = tid >> 5;          // wave id 0..15
    const int lane   = tid & 31;
    const int lanelo = lane & 15;
    const int hi8    = (lane >> 4) * 8;   // M offset for C/D layout, K-select for A/B
    const int sel    = hi8;               // K sub-select (0 or 8)
    const int bbase  = blockIdx.x * 16;   // first batch row of this block
    const int colw   = w * 64;            // first output column of this wave

    v8f acc[4];
    v8f hsum[4];
    float bias[4];
    const unsigned short* pB[4];

#pragma unroll
    for (int t = 0; t < 4; ++t) {
        acc[t]  = (v8f){0.f, 0.f, 0.f, 0.f, 0.f, 0.f, 0.f, 0.f};
        hsum[t] = (v8f){0.f, 0.f, 0.f, 0.f, 0.f, 0.f, 0.f, 0.f};
        bias[t] = bvec[colw + 16 * t + lanelo];
        // B-tile base: lane maps to output column N; row-major W_bf16[N][K]
        pB[t] = Wb + (size_t)(colw + 16 * t + lanelo) * EMB_ + sel;
    }

    const unsigned short* Arow = As + (size_t)lanelo * EMB_ + sel;

    // Software-pipelined token fetch: tok[] holds x[.][i] for current step.
    int tok[8];
#pragma unroll
    for (int j = 0; j < 8; ++j)
        tok[j] = x[(bbase + j + hi8) * S_ + 0];

    for (int i = 0; i < S_; ++i) {
        // ---- embedding row pointers for this step (tokens already in regs) ----
        const float* rowp[8];
#pragma unroll
        for (int j = 0; j < 8; ++j)
            rowp[j] = emb + (size_t)tok[j] * EMB_ + colw + lanelo;

        // ---- fetch next step's tokens and prefetch their embedding rows ----
        if (i + 1 < S_) {
#pragma unroll
            for (int j = 0; j < 8; ++j) {
                tok[j] = x[(bbase + j + hi8) * S_ + i + 1];
                __builtin_prefetch(emb + (size_t)tok[j] * EMB_ + colw + lanelo, 0, 1);
            }
        }

        // ---- stage A = bf16(h_prev + x_i) into LDS (row-major 16 x 1024) ----
#pragma unroll
        for (int t = 0; t < 4; ++t) {
#pragma unroll
            for (int j = 0; j < 8; ++j) {
                float v = acc[t][j] + rowp[j][16 * t];
                As[(size_t)(j + hi8) * EMB_ + colw + 16 * t + lanelo] = f2bf(v);
            }
        }
        __syncthreads();   // A visible to all waves

        // ---- GEMM: h = A(16x1024) x W^T(:, colw..colw+63) ----
#pragma unroll
        for (int t = 0; t < 4; ++t)
            acc[t] = (v8f){0.f, 0.f, 0.f, 0.f, 0.f, 0.f, 0.f, 0.f};

#pragma unroll 4
        for (int kc = 0; kc < 32; ++kc) {
            const int kb = kc * 32;
            Tile16 a;
            a.u[0] = *(const uint4*)(Arow + kb);        // K sel+0..7
            a.u[1] = *(const uint4*)(Arow + kb + 16);   // K sel+16..23
#pragma unroll
            for (int t = 0; t < 4; ++t) {
                Tile16 bm;
                bm.u[0] = *(const uint4*)(pB[t] + kb);
                bm.u[1] = *(const uint4*)(pB[t] + kb + 16);
                acc[t] = __builtin_amdgcn_wmma_f32_16x16x32_bf16(
                    false, a.v, false, bm.v, (short)0, acc[t], false, false);
            }
        }

        // ---- bias + running sum ----
#pragma unroll
        for (int t = 0; t < 4; ++t) {
#pragma unroll
            for (int j = 0; j < 8; ++j) {
                acc[t][j] += bias[t];
                hsum[t][j] += acc[t][j];
            }
        }
        __syncthreads();   // all waves done reading A before next rewrite
    }

    // ---- dump hsum to LDS (overlays A region; safe after barrier) ----
#pragma unroll
    for (int t = 0; t < 4; ++t)
#pragma unroll
        for (int j = 0; j < 8; ++j)
            Hs[(size_t)(j + hi8) * EMB_ + colw + 16 * t + lanelo] = hsum[t][j];
    __syncthreads();

    // ---- classifier head: pred_mean[m][o] = (hsum[m]/S) . Wc[o] + bc[o] ----
    if (tid < 16 * OUT_) {
        int m = tid / OUT_;
        int o = tid % OUT_;
        const float* hr  = Hs + (size_t)m * EMB_;
        const float* wcr = Wc + (size_t)o * EMB_;
        float dot = 0.f;
        for (int n = 0; n < EMB_; ++n) dot += hr[n] * wcr[n];
        predmean[(bbase + m) * OUT_ + o] = dot * (1.0f / (float)S_) + bc[o];
    }
}

// ---------------------------------------------------------------------------
// Kernel 3: cross-entropy (mean) over pred_mean (64 x 10)
// ---------------------------------------------------------------------------
__global__ void loss_kernel(const float* __restrict__ pm,
                            const int*   __restrict__ label,
                            float*       __restrict__ out) {
    __shared__ float r[B_];
    int b = threadIdx.x;
    float p[OUT_];
    float mx = -1e30f;
#pragma unroll
    for (int o = 0; o < OUT_; ++o) {
        p[o] = pm[b * OUT_ + o];
        mx = fmaxf(mx, p[o]);
    }
    float se = 0.f;
#pragma unroll
    for (int o = 0; o < OUT_; ++o) se += __expf(p[o] - mx);
    float lse = __logf(se);
    int lb = label[b];
    r[b] = -(p[lb] - mx - lse);
    __syncthreads();
    if (b == 0) {
        float s = 0.f;
        for (int k = 0; k < B_; ++k) s += r[k];
        out[0] = s * (1.0f / (float)B_);
    }
}

// ---------------------------------------------------------------------------
extern "C" void kernel_launch(void* const* d_in, const int* in_sizes, int n_in,
                              void* d_out, int out_size, void* d_ws, size_t ws_size,
                              hipStream_t stream) {
    (void)in_sizes; (void)n_in; (void)out_size; (void)ws_size;

    const int*   x     = (const int*)d_in[0];
    const int*   label = (const int*)d_in[1];
    const float* emb   = (const float*)d_in[2];
    const float* W     = (const float*)d_in[3];
    const float* bvec  = (const float*)d_in[4];
    const float* Wc    = (const float*)d_in[5];
    const float* bc    = (const float*)d_in[6];

    unsigned short* Wb = (unsigned short*)d_ws;                       // 2 MB bf16 W
    float* pm = (float*)((char*)d_ws + (size_t)EMB_ * EMB_ * sizeof(unsigned short));

    cvt_w_kernel<<<(EMB_ * EMB_) / 256, 256, 0, stream>>>(W, Wb);
    recur_kernel<<<B_ / 16, 512, 0, stream>>>(x, emb, Wb, bvec, Wc, bc, pm);
    loss_kernel<<<1, B_, 0, stream>>>(pm, label, (float*)d_out);
}
